// NEURONSv2TextClassifier_78211354460524
// MI455X (gfx1250) — matrix operand order repaired and checked by reading, not compiled, not run on hardware
//
#include <hip/hip_runtime.h>
#include <math.h>

// ---------------------------------------------------------------------------
// NEURONSv2 text classifier for MI455X (gfx1250, wave32, WMMA).
//
// Pipeline (all bf16 operands, f32 WMMA accumulation):
//   k_embed : Xt[b][d][s]  = bf16(emb[tokens[b,s], d])          (transposed for B-side)
//   k_mask  : Msk[t][s]    = bf16(sigmoid(cos(w*(t-s))) / S)    (Toeplitz, scale folded)
//   k_wsum  : W1s = bf16(w1+fw1), W2s = bf16(w2+fw2)            (row-major [out][in] == B^T)
//   gemm1   : H0[b][t][d]  = sum_s Msk[t][s] * X[b][s][d]        (v_wmma_f32_16x16x32_bf16)
//   gemm2   : H1[b][t][j]  = tanh(sum_d H0*W1s + b1)             (WMMA, tanh epilogue)
//   gemm3   : Hm[b][i]    += tanh(sum_j H1*W2s + b2) / S         (WMMA, fused seq-mean)
//   k_cls   : out = tanh(Hm @ cw1^T + cb1) @ cw2^T + cb2         (tiny, scalar)
//
// Per-wave tile: 32x32 C (2x2 WMMA accs). Operand layouts per ISA 7.12.2:
//   A (16-bit 16x32): lane m=L%16; elems 0..7 @ k0+(hi?8:0), elems 8..15 @ +16.
//   B (16-bit 32x16): lane n=L%16; elems 0..7 @ k0+(hi?8:0), elems 8..15 @ +16 of B^T row n.
//   C (f32 16x16):    lane n=L%16; VGPR r -> row r+8*(L/16).
// ---------------------------------------------------------------------------

#define BATCH 16
#define SEQ   2048
#define DM    512

typedef __attribute__((ext_vector_type(16))) __bf16 bf16x16;
typedef __attribute__((ext_vector_type(8)))  float  f32x8;

union BF16Tile { bf16x16 v; uint4 u[2]; };

// workspace layout (bytes)
#define XT_OFF   0u                         // [B][DM][SEQ] bf16 : 32 MB  (reused as H1)
#define MSK_OFF  33554432u                  // [SEQ][SEQ]  bf16 :  8 MB
#define H0_OFF   41943040u                  // [B][SEQ][DM] bf16: 32 MB
#define W1_OFF   75497472u                  // [DM][DM] bf16   : 0.5 MB
#define W2_OFF   76021760u                  // [DM][DM] bf16   : 0.5 MB
#define HM_OFF   76546048u                  // [B][DM]  f32    : 32 KB

__device__ __forceinline__ unsigned short f2bf(float f) {
  unsigned int u = __float_as_uint(f);
  u += 0x7FFFu + ((u >> 16) & 1u);          // round-to-nearest-even
  return (unsigned short)(u >> 16);
}

// A fragment: rows of row-major A (stride ld, elements), 16x32 bf16 tile at (row m, k0)
__device__ __forceinline__ bf16x16 load_a_tile(const unsigned short* row0,
                                               int ld, int m, int hi, int k0) {
  const unsigned short* p = row0 + (size_t)m * ld + k0 + hi * 8;
  BF16Tile t;
  t.u[0] = *(const uint4*)(p);
  t.u[1] = *(const uint4*)(p + 16);
  return t.v;
}

// B fragment from B^T stored row-major [n][k] (stride ld): elems 0..7 @ k0+(hi?8:0), 8..15 @ +16
__device__ __forceinline__ bf16x16 load_bt_tile(const unsigned short* col0,
                                                int ld, int n, int hi, int k0) {
  const unsigned short* p = col0 + (size_t)n * ld + k0 + hi * 8;
  BF16Tile t;
  t.u[0] = *(const uint4*)(p);
  t.u[1] = *(const uint4*)(p + 16);
  return t.v;
}

#define WMMA_BF16(A, B, C) \
  __builtin_amdgcn_wmma_f32_16x16x32_bf16(false, (A), false, (B), (short)0, (C), false, false)

// ------------------------- prep kernels ------------------------------------

__global__ void k_embed(const int* __restrict__ tokens, const float* __restrict__ emb,
                        unsigned short* __restrict__ Xt) {
  int s = blockIdx.x * 64 + threadIdx.x;
  int d = blockIdx.y * 4 + threadIdx.y;
  int b = blockIdx.z;
  int tok = tokens[(size_t)b * SEQ + s];
  float v = emb[(size_t)tok * DM + d];
  Xt[((size_t)b * DM + d) * SEQ + s] = f2bf(v);   // transposed store, coalesced in s
}

__global__ void k_mask(unsigned short* __restrict__ Msk) {
  int idx = blockIdx.x * 256 + threadIdx.x;
  int t = idx / SEQ, s = idx - t * SEQ;
  const float w = 120.0f * 3.14159265358979323846f / (float)(SEQ - 1);
  float c   = __cosf(w * (float)(t - s));
  float sig = 1.0f / (1.0f + __expf(-c));
  Msk[idx] = f2bf(sig * (1.0f / (float)SEQ));     // fold 1/S into the mask
}

__global__ void k_wsum(const float* __restrict__ a, const float* __restrict__ b,
                       unsigned short* __restrict__ o, int n) {
  int i = blockIdx.x * 256 + threadIdx.x;
  if (i < n) o[i] = f2bf(a[i] + b[i]);
}

__global__ void k_zero(float* __restrict__ p, int n) {
  int i = blockIdx.x * 256 + threadIdx.x;
  if (i < n) p[i] = 0.0f;
}

// ------------------------- GEMM 1: x_att = (mask/S) @ X --------------------

__global__ void __launch_bounds__(256) k_gemm_mask(
    const unsigned short* __restrict__ Msk,   // [SEQ][SEQ]
    const unsigned short* __restrict__ Xt,    // [B][DM][SEQ]  (B^T layout)
    unsigned short* __restrict__ H0) {        // [B][SEQ][DM]
  const int lane = threadIdx.x & 31;
  const int wave = (blockIdx.x * 256 + threadIdx.x) >> 5;
  const int dT = DM / 32, tT = SEQ / 32;      // 16, 64
  int b  = wave / (tT * dT);
  int r  = wave - b * (tT * dT);
  int t0 = (r / dT) * 32;
  int d0 = (r - (r / dT) * dT) * 32;
  const int n  = lane & 15;
  const int hi = lane >> 4;

  const unsigned short* A0 = Msk + (size_t)t0 * SEQ;
  const unsigned short* A1 = Msk + (size_t)(t0 + 16) * SEQ;
  const unsigned short* B0 = Xt + ((size_t)b * DM + d0) * SEQ;
  const unsigned short* B1 = Xt + ((size_t)b * DM + d0 + 16) * SEQ;

  f32x8 c00 = {}, c01 = {}, c10 = {}, c11 = {};
  for (int k0 = 0; k0 < SEQ; k0 += 32) {
    __builtin_prefetch(A0 + (size_t)n * SEQ + k0 + 128, 0, 0);   // global_prefetch_b8
    bf16x16 a0 = load_a_tile(A0, SEQ, n, hi, k0);
    bf16x16 a1 = load_a_tile(A1, SEQ, n, hi, k0);
    bf16x16 b0 = load_bt_tile(B0, SEQ, n, hi, k0);
    bf16x16 b1 = load_bt_tile(B1, SEQ, n, hi, k0);
    c00 = WMMA_BF16(a0, b0, c00);
    c01 = WMMA_BF16(a0, b1, c01);
    c10 = WMMA_BF16(a1, b0, c10);
    c11 = WMMA_BF16(a1, b1, c11);
  }
  unsigned short* out = H0 + (size_t)b * SEQ * DM;
#pragma unroll
  for (int rr = 0; rr < 8; ++rr) {
    int m = rr + hi * 8;
    out[(size_t)(t0 + m) * DM + d0 + n]           = f2bf(c00[rr]);
    out[(size_t)(t0 + m) * DM + d0 + 16 + n]      = f2bf(c01[rr]);
    out[(size_t)(t0 + 16 + m) * DM + d0 + n]      = f2bf(c10[rr]);
    out[(size_t)(t0 + 16 + m) * DM + d0 + 16 + n] = f2bf(c11[rr]);
  }
}

// ------------------------- GEMM 2: h = tanh(H @ W^T + b) -------------------

__global__ void __launch_bounds__(256) k_gemm_ffn(
    const unsigned short* __restrict__ Hin,   // [B][SEQ][DM]
    const unsigned short* __restrict__ W,     // [DM][DM] row-major [out][in] == B^T
    const float* __restrict__ bias,           // [DM]
    unsigned short* __restrict__ Hout) {      // [B][SEQ][DM]
  const int lane = threadIdx.x & 31;
  const int wave = (blockIdx.x * 256 + threadIdx.x) >> 5;
  const int jT = DM / 32, tT = SEQ / 32;
  int b  = wave / (tT * jT);
  int r  = wave - b * (tT * jT);
  int t0 = (r / jT) * 32;
  int j0 = (r - (r / jT) * jT) * 32;
  const int n  = lane & 15;
  const int hi = lane >> 4;

  const unsigned short* A0 = Hin + ((size_t)b * SEQ + t0) * DM;
  const unsigned short* A1 = Hin + ((size_t)b * SEQ + t0 + 16) * DM;
  const unsigned short* B0 = W + (size_t)j0 * DM;
  const unsigned short* B1 = W + (size_t)(j0 + 16) * DM;

  f32x8 c00 = {}, c01 = {}, c10 = {}, c11 = {};
  for (int k0 = 0; k0 < DM; k0 += 32) {
    bf16x16 a0 = load_a_tile(A0, DM, n, hi, k0);
    bf16x16 a1 = load_a_tile(A1, DM, n, hi, k0);
    bf16x16 b0 = load_bt_tile(B0, DM, n, hi, k0);
    bf16x16 b1 = load_bt_tile(B1, DM, n, hi, k0);
    c00 = WMMA_BF16(a0, b0, c00);
    c01 = WMMA_BF16(a0, b1, c01);
    c10 = WMMA_BF16(a1, b0, c10);
    c11 = WMMA_BF16(a1, b1, c11);
  }
  float bi0 = bias[j0 + n];
  float bi1 = bias[j0 + 16 + n];
  unsigned short* out = Hout + (size_t)b * SEQ * DM;
#pragma unroll
  for (int rr = 0; rr < 8; ++rr) {
    int m = rr + hi * 8;
    out[(size_t)(t0 + m) * DM + j0 + n]           = f2bf(tanhf(c00[rr] + bi0));
    out[(size_t)(t0 + m) * DM + j0 + 16 + n]      = f2bf(tanhf(c01[rr] + bi1));
    out[(size_t)(t0 + 16 + m) * DM + j0 + n]      = f2bf(tanhf(c10[rr] + bi0));
    out[(size_t)(t0 + 16 + m) * DM + j0 + 16 + n] = f2bf(tanhf(c11[rr] + bi1));
  }
}

// --------------- GEMM 3: tanh(H @ W^T + b), fused mean over seq ------------

__global__ void __launch_bounds__(256) k_gemm_head(
    const unsigned short* __restrict__ Hin,   // [B][SEQ][DM]
    const unsigned short* __restrict__ W,     // [DM][DM]
    const float* __restrict__ bias,           // [DM]
    float* __restrict__ Hm) {                 // [B][DM] f32, pre-zeroed
  const int lane = threadIdx.x & 31;
  const int wave = (blockIdx.x * 256 + threadIdx.x) >> 5;
  const int iT = DM / 32;                     // 16
  const int CHUNK = 8;                        // 8 t-tiles of 32 rows = 256 rows/wave
  const int nC = SEQ / (32 * CHUNK);          // 8 chunks
  int b  = wave / (iT * nC);
  int r  = wave - b * (iT * nC);
  int i0 = (r / nC) * 32;
  int tc = (r - (r / nC) * nC);
  const int n  = lane & 15;
  const int hi = lane >> 4;

  const unsigned short* B0 = W + (size_t)i0 * DM;
  const unsigned short* B1 = W + (size_t)(i0 + 16) * DM;
  float bi0 = bias[i0 + n];
  float bi1 = bias[i0 + 16 + n];

  float cs0 = 0.0f, cs1 = 0.0f;               // per-lane column partial sums
  for (int tt = 0; tt < CHUNK; ++tt) {
    int t0 = (tc * CHUNK + tt) * 32;
    const unsigned short* A0 = Hin + ((size_t)b * SEQ + t0) * DM;
    const unsigned short* A1 = Hin + ((size_t)b * SEQ + t0 + 16) * DM;
    f32x8 c00 = {}, c01 = {}, c10 = {}, c11 = {};
    for (int k0 = 0; k0 < DM; k0 += 32) {
      bf16x16 a0 = load_a_tile(A0, DM, n, hi, k0);
      bf16x16 a1 = load_a_tile(A1, DM, n, hi, k0);
      bf16x16 b0 = load_bt_tile(B0, DM, n, hi, k0);
      bf16x16 b1 = load_bt_tile(B1, DM, n, hi, k0);
      c00 = WMMA_BF16(a0, b0, c00);
      c01 = WMMA_BF16(a0, b1, c01);
      c10 = WMMA_BF16(a1, b0, c10);
      c11 = WMMA_BF16(a1, b1, c11);
    }
#pragma unroll
    for (int rr = 0; rr < 8; ++rr) {
      cs0 += tanhf(c00[rr] + bi0) + tanhf(c10[rr] + bi0);
      cs1 += tanhf(c01[rr] + bi1) + tanhf(c11[rr] + bi1);
    }
  }
  // combine the two lane halves (rows 0..7+16*rt vs 8..15+16*rt of each column)
  cs0 += __shfl_down(cs0, 16, 32);
  cs1 += __shfl_down(cs1, 16, 32);
  if (lane < 16) {
    atomicAdd(&Hm[(size_t)b * DM + i0 + lane],      cs0 * (1.0f / (float)SEQ));
    atomicAdd(&Hm[(size_t)b * DM + i0 + 16 + lane], cs1 * (1.0f / (float)SEQ));
  }
}

// ------------------------- classifier head (tiny) --------------------------

__global__ void __launch_bounds__(256) k_classifier(
    const float* __restrict__ Hm,             // [B][DM]
    const float* __restrict__ cw1, const float* __restrict__ cb1,
    const float* __restrict__ cw2, const float* __restrict__ cb2,
    float* __restrict__ outp) {               // [B][2]
  __shared__ float hc[BATCH * 256];
  int tid = threadIdx.x;
  for (int idx = tid; idx < BATCH * 256; idx += 256) {
    int b = idx >> 8, i = idx & 255;
    float s = cb1[i];
    const float* hb = Hm + (size_t)b * DM;
    const float* w  = cw1 + (size_t)i * DM;
    for (int d = 0; d < DM; ++d) s += hb[d] * w[d];
    hc[idx] = tanhf(s);
  }
  __syncthreads();
  if (tid < BATCH * 2) {
    int b = tid >> 1, c = tid & 1;
    float s = cb2[c];
    const float* h = hc + b * 256;
    const float* w = cw2 + (size_t)c * 256;
    for (int i = 0; i < 256; ++i) s += h[i] * w[i];
    outp[tid] = s;
  }
}

// ------------------------- launcher ----------------------------------------

extern "C" void kernel_launch(void* const* d_in, const int* in_sizes, int n_in,
                              void* d_out, int out_size, void* d_ws, size_t ws_size,
                              hipStream_t stream) {
  const int*   tokens = (const int*)  d_in[0];
  const float* emb    = (const float*)d_in[1];
  const float* w1     = (const float*)d_in[2];
  const float* b1     = (const float*)d_in[3];
  const float* fw1    = (const float*)d_in[4];
  const float* w2     = (const float*)d_in[5];
  const float* b2     = (const float*)d_in[6];
  const float* fw2    = (const float*)d_in[7];
  const float* cw1    = (const float*)d_in[8];
  const float* cb1    = (const float*)d_in[9];
  const float* cw2    = (const float*)d_in[10];
  const float* cb2    = (const float*)d_in[11];
  float* outp = (float*)d_out;

  char* ws = (char*)d_ws;
  unsigned short* Xt  = (unsigned short*)(ws + XT_OFF);
  unsigned short* Msk = (unsigned short*)(ws + MSK_OFF);
  unsigned short* H0  = (unsigned short*)(ws + H0_OFF);
  unsigned short* W1s = (unsigned short*)(ws + W1_OFF);
  unsigned short* W2s = (unsigned short*)(ws + W2_OFF);
  float*          Hm  = (float*)(ws + HM_OFF);
  unsigned short* H1  = Xt;                   // Xt dead after gemm1 -> reuse

  k_embed<<<dim3(SEQ / 64, DM / 4, BATCH), dim3(64, 4), 0, stream>>>(tokens, emb, Xt);
  k_mask <<<(SEQ * SEQ) / 256, 256, 0, stream>>>(Msk);
  k_wsum <<<(DM * DM) / 256, 256, 0, stream>>>(w1, fw1, W1s, DM * DM);
  k_wsum <<<(DM * DM) / 256, 256, 0, stream>>>(w2, fw2, W2s, DM * DM);
  k_zero <<<(BATCH * DM) / 256, 256, 0, stream>>>(Hm, BATCH * DM);

  const int waves12 = BATCH * (SEQ / 32) * (DM / 32);     // 16384 waves
  k_gemm_mask<<<waves12 / 8, 256, 0, stream>>>(Msk, Xt, H0);
  k_gemm_ffn <<<waves12 / 8, 256, 0, stream>>>(H0, W1s, b1, H1);

  const int waves3 = BATCH * (DM / 32) * 8;               // 2048 waves
  k_gemm_head<<<waves3 / 8, 256, 0, stream>>>(H1, W2s, b2, Hm);

  k_classifier<<<1, 256, 0, stream>>>(Hm, cw1, cb1, cw2, cb2, outp);
}
